// RNN_7361573945507
// MI455X (gfx1250) — matrix operand order repaired and checked by reading, compiled, not run on hardware
//
#include <hip/hip_runtime.h>
#include <math.h>

// ---------------------------------------------------------------------------
// RNN on MI455X (gfx1250, wave32).
//   GEMMs:  split-bf16 WMMA (v_wmma_f32_16x16x32_bf16); weights pre-split to
//           bf16 hi/lo once, then streamed into double-buffered LDS by the
//           Tensor Data Mover (tensor_load_to_lds + s_wait_tensorcnt), with
//           TDM padding producing the conflict-free 80B row stride directly.
//   Scan :  16 WGs (one per head) x 8 waves; state_weight register-resident
//           as split-bf16 B-frags; h circulates through LDS; 12 WMMAs/step;
//           x_t loads software-pipelined one step ahead to hide HBM latency.
// ---------------------------------------------------------------------------

#define BATCH 4
#define SEQ   2048
#define DIM   2048          // INPUT == STATE == OUTPUT size
#define HEADS 16
#define HDIM  128

typedef __attribute__((ext_vector_type(16))) __bf16 v16bf;
typedef __attribute__((ext_vector_type(4)))  __bf16 v4bf;
typedef __attribute__((ext_vector_type(8)))  float  v8f;
typedef __attribute__((ext_vector_type(4)))  float  v4f;
typedef __attribute__((ext_vector_type(4)))  unsigned int v4u;
typedef __attribute__((ext_vector_type(8)))  int    v8i;
typedef __attribute__((ext_vector_type(4)))  int    v4i;

union Frag16 {            // one WMMA A/B operand: 16 bf16 per lane (8 VGPRs)
    v16bf bf;
    v4f   f4[2];
};

// ---- Tensor Data Mover: 2D tile load, 128 rows x 32 bf16, row stride 2048
// elements in memory, 64B rows + 16B LDS padding (-> 80B = LDK rows). -------
__device__ __forceinline__ unsigned lds_off(const void* p) {
    return (unsigned)(unsigned long long)p;     // LDS aperture: addr[31:0]
}

__device__ __forceinline__ void tdm_load_tile(unsigned lds_addr,
                                              const void* gptr) {
    const unsigned long long ga = (unsigned long long)gptr;
    v4u g0;
    g0.x = 1u;                                   // count=1 (valid descriptor)
    g0.y = lds_addr;                             // lds_addr
    g0.z = (unsigned)ga;                         // global_addr[31:0]
    g0.w = (unsigned)((ga >> 32) & 0x01FFFFFFu)  // global_addr[56:32]
         | (2u << 30);                           // type=2 ("image")
    v8i g1;
    g1[0] = (1 << 16)        // data_size = 1 -> 2 bytes
          | (1 << 20)        // pad_enable
          | (3 << 22)        // pad_interval: 16 DWORDs (= 64B row)
          | (3 << 25);       // pad_amount:   4 DWORDs (= 16B pad)
    g1[1] = (int)(2048u << 16);   // tensor_dim0 = 2048 (bits 79:48, low half)
    g1[2] = (int)(128u << 16);    // tensor_dim1 = 128  (bits 111:80, low half)
    g1[3] = (int)(32u << 16);     // tile_dim0 = 32     (bits 127:112)
    g1[4] = 128;                  // tile_dim1 = 128    (bits 143:128)
    g1[5] = 2048;                 // tensor_dim0_stride = 2048 (bits 207:160)
    g1[6] = 0;
    g1[7] = 0;
    const v4i z4 = (v4i){0, 0, 0, 0};
#if __clang_major__ >= 23
    const v8i z8 = (v8i){0, 0, 0, 0, 0, 0, 0, 0};
    __builtin_amdgcn_tensor_load_to_lds(g0, g1, z4, z4, z8, 0);
#else
    __builtin_amdgcn_tensor_load_to_lds(g0, g1, z4, z4, 0);
#endif
}

// ---- weight pre-split: fp32 -> bf16 hi + bf16 lo (residual) ----------------
__global__ __launch_bounds__(256) void split_bf16_kernel(
    const float* __restrict__ src,
    __bf16* __restrict__ hi, __bf16* __restrict__ lo, int n4)
{
    const int i = blockIdx.x * 256 + threadIdx.x;
    if (i >= n4) return;
    const v4f v = ((const v4f*)src)[i];
    v4bf h, l;
    #pragma unroll
    for (int j = 0; j < 4; ++j) {
        const __bf16 hv = (__bf16)v[j];
        h[j] = hv;
        l[j] = (__bf16)(v[j] - (float)hv);
    }
    ((v4bf*)hi)[i] = h;
    ((v4bf*)lo)[i] = l;
}

// -------------------- split-bf16 WMMA GEMM:  C = A * Bw^T + bias ------------
// Tile: 128(M) x 128(N) per WG, K-chunks of 32.  8 waves; each wave owns a
// 32x64 sub-tile = 2x4 WMMA 16x16 D-tiles.  A (fp32 activations) split on
// the fly into LDS; B (pre-split weights) TDM-loaded, double-buffered.
// LDS rows padded to 40 bf16 (80B) -> conflict-free ds_load_b128 frag reads.

#define LDK 40

__global__ __launch_bounds__(256) void wmma_gemm_tdm(
    const float*  __restrict__ A,      // [M][2048] fp32
    const __bf16* __restrict__ Bhi_g,  // [2048][2048] bf16 hi
    const __bf16* __restrict__ Blo_g,  // [2048][2048] bf16 lo
    const float*  __restrict__ bias,   // [2048] or nullptr
    float* __restrict__ C)             // [M][2048]
{
    __shared__ alignas(16) __bf16 Ahi[128][LDK];
    __shared__ alignas(16) __bf16 Alo[128][LDK];
    __shared__ alignas(16) __bf16 Bbh[2][128][LDK];   // TDM destinations
    __shared__ alignas(16) __bf16 Bbl[2][128][LDK];

    const int tid    = threadIdx.x;
    const int lane   = tid & 31;
    const int wave   = tid >> 5;          // 0..7
    const int warp_m = wave & 3;          // 4 row-groups of 32
    const int warp_n = wave >> 2;         // 2 col-groups of 64
    const int half   = lane >> 4;         // lane half selects K sub-range
    const int lr     = lane & 15;         // row (A:M, B:N) within 16-tile

    const int m0 = blockIdx.y * 128;
    const int n0 = blockIdx.x * 128;

    v8f acc[2][4];
    #pragma unroll
    for (int mi = 0; mi < 2; ++mi)
        #pragma unroll
        for (int ni = 0; ni < 4; ++ni)
            acc[mi][ni] = (v8f){0.f,0.f,0.f,0.f,0.f,0.f,0.f,0.f};

    // prologue: start DMA of B tiles for chunk 0 into buffer 0
    if (wave == 0) {
        tdm_load_tile(lds_off(&Bbh[0][0][0]), Bhi_g + (size_t)n0 * DIM);
        tdm_load_tile(lds_off(&Bbl[0][0][0]), Blo_g + (size_t)n0 * DIM);
    }

    int cur = 0;
    for (int kc = 0; kc < DIM; kc += 32, cur ^= 1) {
        __syncthreads();   // prior compute done: frees A bufs + Bbuf[cur^1]

        // ---- stage 128x32 fp32 A tile into LDS as bf16 hi/lo --------------
        // 1024 float4 / 256 threads = 4 each (overlaps in-flight TDM).
        #pragma unroll
        for (int i = tid; i < 128 * 8; i += 256) {
            const int row = i >> 3;
            const int c4  = (i & 7) << 2;                 // 0,4,...,28
            const float* pa = A + (size_t)(m0 + row) * DIM + kc + c4;
            v4f av = *(const v4f*)pa;
            if (kc + 32 < DIM)
                __builtin_prefetch(pa + 32, 0, 3);        // WGP-scope prefetch
            v4bf ah, al;
            #pragma unroll
            for (int j = 0; j < 4; ++j) {
                const __bf16 h = (__bf16)av[j];
                ah[j] = h;  al[j] = (__bf16)(av[j] - (float)h);
            }
            *(v4bf*)&Ahi[row][c4] = ah;  *(v4bf*)&Alo[row][c4] = al;
        }

        // ---- TDM: issue next chunk's B tiles, retire current --------------
        if (wave == 0) {
            if (kc + 32 < DIM) {
                const size_t go = (size_t)n0 * DIM + kc + 32;
                tdm_load_tile(lds_off(&Bbh[cur ^ 1][0][0]), Bhi_g + go);
                tdm_load_tile(lds_off(&Bbl[cur ^ 1][0][0]), Blo_g + go);
                __builtin_amdgcn_s_wait_tensorcnt((short)2);  // cur complete
            } else {
                __builtin_amdgcn_s_wait_tensorcnt((short)0);
            }
        }
        __syncthreads();   // A staged + B[cur] DMA complete, visible to all

        // ---- load fragments per ISA 7.12.2 wave32 layouts -----------------
        Frag16 fah[2], fal[2], fbh[4], fbl[4];
        #pragma unroll
        for (int mi = 0; mi < 2; ++mi) {
            const int m = warp_m * 32 + mi * 16 + lr;
            fah[mi].f4[0] = *(const v4f*)&Ahi[m][half * 8];
            fah[mi].f4[1] = *(const v4f*)&Ahi[m][16 + half * 8];
            fal[mi].f4[0] = *(const v4f*)&Alo[m][half * 8];
            fal[mi].f4[1] = *(const v4f*)&Alo[m][16 + half * 8];
        }
        #pragma unroll
        for (int ni = 0; ni < 4; ++ni) {
            const int n = warp_n * 64 + ni * 16 + lr;
            fbh[ni].f4[0] = *(const v4f*)&Bbh[cur][n][half * 8];
            fbh[ni].f4[1] = *(const v4f*)&Bbh[cur][n][16 + half * 8];
            fbl[ni].f4[0] = *(const v4f*)&Bbl[cur][n][half * 8];
            fbl[ni].f4[1] = *(const v4f*)&Bbl[cur][n][16 + half * 8];
        }

        // ---- 3-product split-bf16 accumulation ----------------------------
        #pragma unroll
        for (int mi = 0; mi < 2; ++mi)
            #pragma unroll
            for (int ni = 0; ni < 4; ++ni) {
                acc[mi][ni] = __builtin_amdgcn_wmma_f32_16x16x32_bf16(
                    false, fah[mi].bf, false, fbh[ni].bf,
                    (short)0, acc[mi][ni], false, false);
                acc[mi][ni] = __builtin_amdgcn_wmma_f32_16x16x32_bf16(
                    false, fah[mi].bf, false, fbl[ni].bf,
                    (short)0, acc[mi][ni], false, false);
                acc[mi][ni] = __builtin_amdgcn_wmma_f32_16x16x32_bf16(
                    false, fal[mi].bf, false, fbh[ni].bf,
                    (short)0, acc[mi][ni], false, false);
            }
    }

    // ---- epilogue: D layout = VGPR r -> M=r (lanes 0-15) / M=r+8 (16-31),
    //                N = lane % 16 ------------------------------------------
    #pragma unroll
    for (int mi = 0; mi < 2; ++mi)
        #pragma unroll
        for (int ni = 0; ni < 4; ++ni) {
            const int col  = n0 + warp_n * 64 + ni * 16 + lr;
            const float bv = bias ? bias[col] : 0.f;
            const int rb   = m0 + warp_m * 32 + mi * 16 + half * 8;
            #pragma unroll
            for (int r = 0; r < 8; ++r)
                C[(size_t)(rb + r) * DIM + col] = acc[mi][ni][r] + bv;
        }
}

// -------------------- WMMA sequential scan: h = tanh(h @ W_h + x_t) ---------
// Grid = 16 heads; 256 threads = 8 waves.  Wave w owns N-slice [w*16, w*16+16)
// of W_h, register-resident as split-bf16 B-frags (4 K-chunks x hi/lo = 64
// VGPRs).  h is M=16 (batches 0..3 live, 4..15 zero) x K=128 in LDS as
// split-bf16, rows padded to 136 (lane stride = 4 banks -> conflict-free
// ds_load_b128 A-frag reads).  12 WMMAs/step in 3 independent chains.
// x_t+1 global loads issued at the top of step t (register double buffer) so
// HBM latency hides under the step's LDS/WMMA/tanh/barrier work.

#define HPAD 136

__global__ __launch_bounds__(256) void rnn_scan_wmma(
    const float* __restrict__ x,        // [B][S][DIM]
    const float* __restrict__ h0,       // [B][DIM]
    const float* __restrict__ Wstate,   // [HEADS][HDIM][HDIM]
    float* __restrict__ states,         // [B][S][DIM]
    float* __restrict__ final_state)    // [B][DIM]
{
    __shared__ alignas(16) __bf16 h_hi[16][HPAD];
    __shared__ alignas(16) __bf16 h_lo[16][HPAD];

    const int head = blockIdx.x;
    const int tid  = threadIdx.x;
    const int lane = tid & 31;
    const int wave = tid >> 5;          // n-tile owner
    const int half = lane >> 4;
    const int lr   = lane & 15;
    const int n0   = wave * 16;

    // zero all 16 rows (rows 4..15 stay zero -> valid M=16 WMMA A tile)
    for (int i = tid; i < 16 * HPAD; i += 256) {
        (&h_hi[0][0])[i] = (__bf16)0.f;
        (&h_lo[0][0])[i] = (__bf16)0.f;
    }
    __syncthreads();
    // rows 0..3 <- input_state, split hi/lo
    for (int i = tid; i < BATCH * HDIM; i += 256) {
        const int b = i >> 7, e = i & 127;
        const float v = h0[(size_t)b * DIM + head * HDIM + e];
        const __bf16 hi = (__bf16)v;
        h_hi[b][e] = hi;
        h_lo[b][e] = (__bf16)(v - (float)hi);
    }

    // ---- register-resident W slice: B-frags per ISA layout ----------------
    Frag16 wbh[4], wbl[4];
    #pragma unroll
    for (int kc = 0; kc < 4; ++kc) {
        #pragma unroll
        for (int r = 0; r < 8; ++r)
            #pragma unroll
            for (int j = 0; j < 2; ++j) {
                const int d = kc * 32 + ((r < 4) ? half * 8 + 2 * r + j : 16 + half * 8 + 2 * r + j - 8);
                const float v =
                    Wstate[((size_t)head * HDIM + d) * HDIM + n0 + lr];
                const __bf16 hi = (__bf16)v;
                wbh[kc].bf[2 * r + j] = hi;
                wbl[kc].bf[2 * r + j] = (__bf16)(v - (float)hi);
            }
    }
    __syncthreads();

    const v8f vzero = (v8f){0.f,0.f,0.f,0.f,0.f,0.f,0.f,0.f};
    const size_t xbase = (size_t)head * HDIM + n0 + lr;   // per-lane column
    float last[4] = {0.f, 0.f, 0.f, 0.f};

    // preload x for t = 0
    float xv[4];
    #pragma unroll
    for (int r = 0; r < 4; ++r)
        xv[r] = x[(size_t)r * SEQ * DIM + xbase];

    for (int t = 0; t < SEQ; ++t) {
        // ---- issue next step's x loads first (latency hiding) -------------
        float xnext[4] = {0.f, 0.f, 0.f, 0.f};
        if (t + 1 < SEQ) {
            #pragma unroll
            for (int r = 0; r < 4; ++r)
                xnext[r] = x[((size_t)r * SEQ + t + 1) * DIM + xbase];
        }

        // ---- A-frags (h) from LDS: lane lr -> M=lr ------------------------
        Frag16 fah[4], fal[4];
        #pragma unroll
        for (int kc = 0; kc < 4; ++kc) {
            fah[kc].f4[0] = *(const v4f*)&h_hi[lr][kc * 32 + half * 8];
            fah[kc].f4[1] = *(const v4f*)&h_hi[lr][kc * 32 + 16 + half * 8];
            fal[kc].f4[0] = *(const v4f*)&h_lo[lr][kc * 32 + half * 8];
            fal[kc].f4[1] = *(const v4f*)&h_lo[lr][kc * 32 + 16 + half * 8];
        }

        // ---- 3 independent split-bf16 accumulation chains -----------------
        v8f a0 = vzero, a1 = vzero, a2 = vzero;
        #pragma unroll
        for (int kc = 0; kc < 4; ++kc) {
            a0 = __builtin_amdgcn_wmma_f32_16x16x32_bf16(
                false, fah[kc].bf, false, wbh[kc].bf, (short)0, a0, false, false);
            a1 = __builtin_amdgcn_wmma_f32_16x16x32_bf16(
                false, fah[kc].bf, false, wbl[kc].bf, (short)0, a1, false, false);
            a2 = __builtin_amdgcn_wmma_f32_16x16x32_bf16(
                false, fal[kc].bf, false, wbh[kc].bf, (short)0, a2, false, false);
        }

        // ---- y = tanh(D + x_t): live rows are VGPR 0..3, lanes 0..15 ------
        float nh[4];
        #pragma unroll
        for (int r = 0; r < 4; ++r)
            nh[r] = tanhf(a0[r] + a1[r] + a2[r] + xv[r]);

        __syncthreads();                    // all A-frag reads done
        if (lane < 16) {
            #pragma unroll
            for (int r = 0; r < 4; ++r) {
                states[((size_t)r * SEQ + t) * DIM + xbase] = nh[r];
                const __bf16 hi = (__bf16)nh[r];
                h_hi[r][n0 + lr] = hi;
                h_lo[r][n0 + lr] = (__bf16)(nh[r] - (float)hi);
                last[r] = nh[r];
            }
        }
        __syncthreads();                    // new h visible before next reads

        #pragma unroll
        for (int r = 0; r < 4; ++r)
            xv[r] = xnext[r];
    }

    if (lane < 16) {
        #pragma unroll
        for (int r = 0; r < 4; ++r)
            final_state[(size_t)r * DIM + xbase] = last[r];
    }
}

// ---------------------------------------------------------------------------

extern "C" void kernel_launch(void* const* d_in, const int* in_sizes, int n_in,
                              void* d_out, int out_size, void* d_ws, size_t ws_size,
                              hipStream_t stream) {
    const float* input       = (const float*)d_in[0];  // [B][S][DIM]
    const float* input_state = (const float*)d_in[1];  // [B][DIM]
    const float* W_in        = (const float*)d_in[2];  // [DIM][DIM]
    const float* b_in        = (const float*)d_in[3];  // [DIM]
    const float* state_w     = (const float*)d_in[4];  // [HEADS][HDIM][HDIM]
    const float* W_out       = (const float*)d_in[5];  // [DIM][DIM]

    const size_t bsd = (size_t)BATCH * SEQ * DIM;      // 16.7M floats
    float* out         = (float*)d_out;                // [B][S][DIM]
    float* final_state = out + bsd;                    // [B][DIM]

    float*  xbuf   = (float*)d_ws;                     // 67 MB
    float*  states = xbuf + bsd;                       // 67 MB
    __bf16* wihi   = (__bf16*)(states + bsd);          // 8.4 MB each
    __bf16* wilo   = wihi + (size_t)DIM * DIM;
    __bf16* wohi   = wilo + (size_t)DIM * DIM;
    __bf16* wolo   = wohi + (size_t)DIM * DIM;

    const int n4 = DIM * DIM / 4;                      // float4s per weight
    const dim3 gemm_grid(DIM / 128, (BATCH * SEQ) / 128);   // (16, 64)

    // pre-split weights into persistent bf16 hi/lo (TDM sources)
    split_bf16_kernel<<<n4 / 256, 256, 0, stream>>>(W_in,  wihi, wilo, n4);
    split_bf16_kernel<<<n4 / 256, 256, 0, stream>>>(W_out, wohi, wolo, n4);

    // x = input @ W_in^T + b_in
    wmma_gemm_tdm<<<gemm_grid, 256, 0, stream>>>(input, wihi, wilo, b_in, xbuf);
    // sequential tanh recurrence on the WMMA units
    rnn_scan_wmma<<<HEADS, 256, 0, stream>>>(xbuf, input_state, state_w,
                                             states, final_state);
    // out = states @ W_out^T
    wmma_gemm_tdm<<<gemm_grid, 256, 0, stream>>>(states, wohi, wolo, nullptr, out);
}
